// Mamba_head_78254304133372
// MI455X (gfx1250) — matrix operand order repaired and checked by reading, not compiled, not run on hardware
//
#include <hip/hip_runtime.h>
#include <hip/hip_bf16.h>

// ---------------------------------------------------------------------------
// Types for CDNA5 WMMA (wave32)
// ---------------------------------------------------------------------------
typedef __attribute__((ext_vector_type(16))) _Float16 v16h;
typedef __attribute__((ext_vector_type(8)))  _Float16 v8h;
typedef __attribute__((ext_vector_type(8)))  float    v8f;

union AFrag { v16h v; struct { v8h lo; v8h hi; } p; };

#define BATCH   4
#define SEQL    1024
#define DMODEL  1024
#define DINNER  1024
#define DSTATE  16
#define DTRANK  64
#define DCONV   4
#define NTOK    (BATCH * SEQL)          // 4096
#define XDBLC   (DTRANK + 2 * DSTATE)   // 96

// ---------------------------------------------------------------------------
// f32 -> f16 convert
// ---------------------------------------------------------------------------
__global__ void __launch_bounds__(256) cvt_f16_kernel(const float* __restrict__ in,
                                                      _Float16* __restrict__ out, int n) {
    int i = blockIdx.x * 256 + threadIdx.x;
    if (i < n) out[i] = (_Float16)in[i];
}

// ---------------------------------------------------------------------------
// LayerNorm over D_MODEL, output f16 (for WMMA A operand)
// ---------------------------------------------------------------------------
__global__ void __launch_bounds__(256) ln_kernel(const float* __restrict__ x,
                                                 const float* __restrict__ g,
                                                 const float* __restrict__ b,
                                                 _Float16* __restrict__ out) {
    const int tok = blockIdx.x;
    const int tid = threadIdx.x;
    const float* xr = x + (size_t)tok * DMODEL;
    float s = 0.f, s2 = 0.f;
    for (int i = tid; i < DMODEL; i += 256) { float v = xr[i]; s += v; s2 += v * v; }
    __shared__ float sb[256];
    sb[tid] = s; __syncthreads();
    for (int st = 128; st > 0; st >>= 1) { if (tid < st) sb[tid] += sb[tid + st]; __syncthreads(); }
    float S = sb[0]; __syncthreads();
    sb[tid] = s2; __syncthreads();
    for (int st = 128; st > 0; st >>= 1) { if (tid < st) sb[tid] += sb[tid + st]; __syncthreads(); }
    float S2 = sb[0];
    float mu  = S * (1.f / DMODEL);
    float var = S2 * (1.f / DMODEL) - mu * mu;
    float inv = rsqrtf(var + 1e-5f);
    _Float16* orow = out + (size_t)tok * DMODEL;
    for (int i = tid; i < DMODEL; i += 256)
        orow[i] = (_Float16)((xr[i] - mu) * inv * g[i] + b[i]);
}

// ---------------------------------------------------------------------------
// Register-blocked WMMA GEMM: C[M,N] = A[M,K](f16,row-major) * W[N,K]^T
// Each wave computes a 32x64 macro-tile (2 M-tiles x 4 N-tiles) so every A
// fragment is reused 4x and every B fragment 2x (8 v_wmma per 12 b128 loads).
// mode 0: store f32   mode 1: store f16   mode 2: f32 + bias[n] + resid[m*N+n]
// Requires: M % 32 == 0, N % 16 == 0, K % 32 == 0. All guards wave-uniform.
// ---------------------------------------------------------------------------
__global__ void __launch_bounds__(128)
gemm_wmma_kernel(const _Float16* __restrict__ A, const _Float16* __restrict__ W,
                 const float* __restrict__ bias, const float* __restrict__ resid,
                 void* __restrict__ out, int M, int N, int K, int mode) {
    const int wave = threadIdx.x >> 5;
    const int lane = threadIdx.x & 31;
    const int tiles_n = N >> 4;
    const int macro_n = (tiles_n + 3) >> 2;
    const int macro_m = M >> 5;
    const int macro = blockIdx.x * 4 + wave;
    if (macro >= macro_m * macro_n) return;       // wave-uniform exit
    const int mm = macro / macro_n;
    const int mn = macro - mm * macro_n;
    const int half = lane >> 4;
    const int lrow = lane & 15;

    const _Float16* a_ptr[2];
#pragma unroll
    for (int i = 0; i < 2; ++i)
        a_ptr[i] = A + (size_t)(mm * 32 + i * 16 + lrow) * K + half * 8;
    const _Float16* b_ptr[4];
#pragma unroll
    for (int j = 0; j < 4; ++j) {
        int rw = mn * 64 + j * 16 + lrow;
        if (rw > N - 1) rw = N - 1;               // clamp: safe read, result discarded
        b_ptr[j] = W + (size_t)rw * K + half * 8;
    }

    v8f acc[2][4];
#pragma unroll
    for (int i = 0; i < 2; ++i)
#pragma unroll
        for (int j = 0; j < 4; ++j) acc[i][j] = {};

    for (int k0 = 0; k0 < K; k0 += 32) {
        AFrag a[2];
#pragma unroll
        for (int i = 0; i < 2; ++i) {
            a[i].p.lo = *(const v8h*)(a_ptr[i] + k0);
            a[i].p.hi = *(const v8h*)(a_ptr[i] + k0 + 16);
        }
        v16h bv[4];
#pragma unroll
        for (int j = 0; j < 4; ++j) { AFrag t; t.p.lo = *(const v8h*)(b_ptr[j] + k0); t.p.hi = *(const v8h*)(b_ptr[j] + k0 + 16); bv[j] = t.v; }
#pragma unroll
        for (int i = 0; i < 2; ++i)
#pragma unroll
            for (int j = 0; j < 4; ++j)
                acc[i][j] = __builtin_amdgcn_wmma_f32_16x16x32_f16(
                    false, a[i].v, false, bv[j], (short)0, acc[i][j], false, false);
    }

#pragma unroll
    for (int i = 0; i < 2; ++i)
#pragma unroll
        for (int j = 0; j < 4; ++j) {
            const int tn = mn * 4 + j;
            if (tn >= tiles_n) continue;          // skip clamped N-tiles
            const int col = tn * 16 + lrow;
#pragma unroll
            for (int r = 0; r < 8; ++r) {
                const int row = mm * 32 + i * 16 + r + half * 8;
                const size_t off = (size_t)row * N + col;
                float v = acc[i][j][r];
                if (mode == 1) {
                    ((_Float16*)out)[off] = (_Float16)v;
                } else {
                    if (mode == 2) v += bias[col] + resid[off];
                    ((float*)out)[off] = v;
                }
            }
        }
}

// ---------------------------------------------------------------------------
// Causal depthwise conv (k=4) + bias + SiLU for one branch direction.
// dir=0: branch time == original time; dir=1: branch reads reversed sequence.
// ---------------------------------------------------------------------------
__global__ void __launch_bounds__(256)
conv_silu_kernel(const float* __restrict__ xz, const float* __restrict__ cw,
                 const float* __restrict__ cb, float* __restrict__ u,
                 _Float16* __restrict__ u16, int dir) {
    const int idx = blockIdx.x * 256 + threadIdx.x;   // over NTOK*DINNER
    const int d   = idx & (DINNER - 1);
    const int tok = idx >> 10;
    const int b   = tok >> 10;
    const int t   = tok & (SEQL - 1);
    float acc = cb[d];
#pragma unroll
    for (int k = 0; k < DCONV; ++k) {
        const int ts = t - (DCONV - 1) + k;
        if (ts >= 0) {
            const int lo = dir ? (SEQL - 1 - ts) : ts;
            acc += cw[d * DCONV + k] * xz[((size_t)(b * SEQL + lo)) * (2 * DINNER) + d];
        }
    }
    const float s = acc / (1.f + __expf(-acc));      // SiLU
    u[idx] = s;
    u16[idx] = (_Float16)s;
}

// ---------------------------------------------------------------------------
// Pack first DTRANK columns of x_dbl [NTOK, 96] into f16 [NTOK, 64]
// ---------------------------------------------------------------------------
__global__ void __launch_bounds__(256)
pack_dt_kernel(const float* __restrict__ xdbl, _Float16* __restrict__ out) {
    const int i = blockIdx.x * 256 + threadIdx.x;    // over NTOK*DTRANK
    const int r = i >> 6;
    const int c = i & 63;
    out[i] = (_Float16)xdbl[(size_t)r * XDBLC + c];
}

// ---------------------------------------------------------------------------
// Selective scan for one branch. One thread per (b,d); h[16] in registers;
// B/C vectors staged through LDS once per timestep. 64-thread blocks spread
// the 4096 independent (b,d) scans across 64 blocks / many WGPs.
// dir=0: y[orig_t] = val ; dir=1: y[orig_t] += val (orig_t = L-1-t).
// ---------------------------------------------------------------------------
__global__ void __launch_bounds__(64)
scan_kernel(const float* __restrict__ u, const float* __restrict__ dtbuf,
            const float* __restrict__ xdbl, const float* __restrict__ xz,
            const float* __restrict__ A_log, const float* __restrict__ dt_bias,
            const float* __restrict__ Dp, float* __restrict__ y, int dir) {
    const int b = blockIdx.y;
    const int d = blockIdx.x * 64 + threadIdx.x;
    float A[DSTATE];
#pragma unroll
    for (int n = 0; n < DSTATE; ++n) A[n] = -__expf(A_log[d * DSTATE + n]);
    const float dtb = dt_bias[d];
    const float Dd  = Dp[d];
    float h[DSTATE];
#pragma unroll
    for (int n = 0; n < DSTATE; ++n) h[n] = 0.f;

    __shared__ float sBC[2 * DSTATE];
    for (int t = 0; t < SEQL; ++t) {
        const int tok = b * SEQL + t;                 // branch-local token
        if (threadIdx.x < 2 * DSTATE)
            sBC[threadIdx.x] = xdbl[(size_t)tok * XDBLC + DTRANK + threadIdx.x];
        __syncthreads();

        float dtv = dtbuf[(size_t)tok * DINNER + d] + dtb;
        float delta = (dtv > 20.f) ? dtv : log1pf(__expf(dtv));   // softplus
        float uv = u[(size_t)tok * DINNER + d];
        float du = delta * uv;
        float yv = 0.f;
#pragma unroll
        for (int n = 0; n < DSTATE; ++n) {
            h[n] = h[n] * __expf(delta * A[n]) + du * sBC[n];
            yv += h[n] * sBC[DSTATE + n];
        }
        __syncthreads();

        const int lo = dir ? (SEQL - 1 - t) : t;      // original time
        const float z = xz[((size_t)(b * SEQL + lo)) * (2 * DINNER) + DINNER + d];
        const float g = z / (1.f + __expf(-z));       // SiLU(z)
        const float val = (yv + Dd * uv) * g;
        const size_t oi = ((size_t)(b * SEQL + lo)) * DINNER + d;
        if (dir) y[oi] += val; else y[oi] = val;
    }
}

// ---------------------------------------------------------------------------
// Host-side launcher
// ---------------------------------------------------------------------------
extern "C" void kernel_launch(void* const* d_in, const int* in_sizes, int n_in,
                              void* d_out, int out_size, void* d_ws, size_t ws_size,
                              hipStream_t stream) {
    (void)in_sizes; (void)n_in; (void)out_size; (void)ws_size;

    const float* hidden = (const float*)d_in[0];
    const float* ln_g   = (const float*)d_in[1];
    const float* ln_b   = (const float*)d_in[2];
    const float* w_in   = (const float*)d_in[3];
    const float* conv_w[2]  = {(const float*)d_in[4],  (const float*)d_in[11]};
    const float* conv_b[2]  = {(const float*)d_in[5],  (const float*)d_in[12]};
    const float* xp_w[2]    = {(const float*)d_in[6],  (const float*)d_in[13]};
    const float* dt_w[2]    = {(const float*)d_in[7],  (const float*)d_in[14]};
    const float* dt_b[2]    = {(const float*)d_in[8],  (const float*)d_in[15]};
    const float* A_log[2]   = {(const float*)d_in[9],  (const float*)d_in[16]};
    const float* Dp[2]      = {(const float*)d_in[10], (const float*)d_in[17]};
    const float* w_out  = (const float*)d_in[18];
    const float* w_fc   = (const float*)d_in[19];
    const float* fc_b   = (const float*)d_in[20];
    float* out = (float*)d_out;

    // ---- carve workspace ----
    char* ws = (char*)d_ws;
    size_t off = 0;
    auto alloc = [&](size_t bytes) -> void* {
        void* p = ws + off;
        off = (off + bytes + 255) & ~(size_t)255;
        return p;
    };
    _Float16* hs16     = (_Float16*)alloc((size_t)NTOK * DMODEL * 2);
    _Float16* w_in16   = (_Float16*)alloc((size_t)2 * DINNER * DMODEL * 2);
    _Float16* w_out16  = (_Float16*)alloc((size_t)DMODEL * DINNER * 2);
    _Float16* w_fc16   = (_Float16*)alloc((size_t)DMODEL * DMODEL * 2);
    _Float16* w_xp16[2], *w_dt16[2];
    for (int s = 0; s < 2; ++s) {
        w_xp16[s] = (_Float16*)alloc((size_t)XDBLC * DINNER * 2);
        w_dt16[s] = (_Float16*)alloc((size_t)DINNER * DTRANK * 2);
    }
    float*    xz     = (float*)   alloc((size_t)NTOK * 2 * DINNER * 4);
    float*    u32    = (float*)   alloc((size_t)NTOK * DINNER * 4);
    _Float16* u16    = (_Float16*)alloc((size_t)NTOK * DINNER * 2);
    float*    xdbl   = (float*)   alloc((size_t)NTOK * XDBLC * 4);
    _Float16* xdbl16 = (_Float16*)alloc((size_t)NTOK * DTRANK * 2);
    float*    dtbuf  = (float*)   alloc((size_t)NTOK * DINNER * 4);
    float*    ysum   = (float*)   alloc((size_t)NTOK * DINNER * 4);
    _Float16* y16    = (_Float16*)alloc((size_t)NTOK * DINNER * 2);
    _Float16* o1_16  = (_Float16*)alloc((size_t)NTOK * DMODEL * 2);

    auto cvt = [&](const float* src, _Float16* dst, int n) {
        cvt_f16_kernel<<<(n + 255) / 256, 256, 0, stream>>>(src, dst, n);
    };
    auto gemm = [&](const _Float16* A, const _Float16* W, const float* bias,
                    const float* resid, void* o, int M, int N, int K, int mode) {
        int macro_total = (M / 32) * (((N / 16) + 3) / 4);
        gemm_wmma_kernel<<<(macro_total + 3) / 4, 128, 0, stream>>>(A, W, bias, resid, o,
                                                                    M, N, K, mode);
    };

    // ---- weight conversion (f32 -> f16) ----
    cvt(w_in,  w_in16,  2 * DINNER * DMODEL);
    cvt(w_out, w_out16, DMODEL * DINNER);
    cvt(w_fc,  w_fc16,  DMODEL * DMODEL);
    for (int s = 0; s < 2; ++s) {
        cvt(xp_w[s], w_xp16[s], XDBLC * DINNER);
        cvt(dt_w[s], w_dt16[s], DINNER * DTRANK);
    }

    // ---- LayerNorm -> f16 ----
    ln_kernel<<<NTOK, 256, 0, stream>>>(hidden, ln_g, ln_b, hs16);

    // ---- in_proj: xz[NTOK, 2048] = hs16 @ w_in^T ----
    gemm(hs16, w_in16, nullptr, nullptr, xz, NTOK, 2 * DINNER, DMODEL, 0);

    // ---- two SSM branches (sequential, reusing buffers) ----
    const int ew = NTOK * DINNER;
    for (int s = 0; s < 2; ++s) {
        conv_silu_kernel<<<ew / 256, 256, 0, stream>>>(xz, conv_w[s], conv_b[s], u32, u16, s);
        gemm(u16, w_xp16[s], nullptr, nullptr, xdbl, NTOK, XDBLC, DINNER, 0);
        pack_dt_kernel<<<(NTOK * DTRANK) / 256, 256, 0, stream>>>(xdbl, xdbl16);
        gemm(xdbl16, w_dt16[s], nullptr, nullptr, dtbuf, NTOK, DINNER, DTRANK, 0);
        scan_kernel<<<dim3(DINNER / 64, BATCH), 64, 0, stream>>>(
            u32, dtbuf, xdbl, xz, A_log[s], dt_b[s], Dp[s], ysum, s);
    }

    // ---- output projections ----
    cvt(ysum, y16, ew);
    gemm(y16, w_out16, nullptr, nullptr, o1_16, NTOK, DMODEL, DINNER, 1);
    gemm(o1_16, w_fc16, fc_b, hidden, out, NTOK, DMODEL, DMODEL, 2);
}
